// GRUModel_87909390614969
// MI455X (gfx1250) — compile-verified, matrix operand order checked by reading, not run on hardware
//
#include <hip/hip_runtime.h>

typedef __attribute__((ext_vector_type(16))) _Float16 v16h;
typedef __attribute__((ext_vector_type(8)))  _Float16 v8h;
typedef __attribute__((ext_vector_type(8)))  float    v8f;

#define SEQ     96
#define CH      2
#define HID     64
#define G3      192          // 3*HID
#define NT      12           // N tiles of 16 over 192
#define KC      2            // K chunks of 32 over 64
#define PRE     12
#define NODES   207
#define BATCH   64
#define ROWS    (BATCH*NODES)   // 13248 = 828 * 16
#define WAVES   4
#define TPB     (WAVES*32)
#define FRAG_H  (32*16)      // f16 elements in one 32x16 B-fragment slot

struct __align__(64) Lds {
  _Float16 Bhh0[NT*KC*FRAG_H];   // 24 KB
  _Float16 Bih1[NT*KC*FRAG_H];   // 24 KB
  _Float16 Bhh1[NT*KC*FRAG_H];   // 24 KB
  _Float16 Bih0[NT*FRAG_H];      // 12 KB: K=0..1 = W_ih0, K=2 = fused biases
  _Float16 h0t[WAVES][16*HID];   // per-wave layer-0 hidden tile, f16 row-major
  _Float16 h1t[WAVES][16*HID];   // per-wave layer-1 hidden tile
};

__device__ __forceinline__ v8f wmma16(v16h a, v16h b, v8f c) {
  return __builtin_amdgcn_wmma_f32_16x16x32_f16(
      false, a, false, b, (short)0, c, false, false);
}

__device__ __forceinline__ v16h ldsB(const _Float16* p) {
  return *(const v16h*)p;   // 32B-aligned per-lane fragment slice
}

// A-fragment (16x32 f16, K-chunk kc) from row-major 16x64 f16 LDS tile.
__device__ __forceinline__ v16h loadA(const _Float16* hb, int kc, int lh, int hi) {
  const _Float16* p = hb + lh*HID + kc*32 + hi*8;
  v8h lo = *(const v8h*)p;
  v8h h8 = *(const v8h*)(p + 16);
  v16h a;
#pragma unroll
  for (int i = 0; i < 8; ++i) { a[i] = lo[i]; a[i+8] = h8[i]; }
  return a;
}

// ---- native CDNA5 tanh (TRANS32 op) ----
#if __has_builtin(__builtin_amdgcn_tanhf)
__device__ __forceinline__ float fast_tanh(float x) { return __builtin_amdgcn_tanhf(x); }
#else
__device__ __forceinline__ float fast_tanh(float x) {
  float r;
  // trailing v_nop: 1-op gap required after a TRANS op before its output is
  // consumed (ISA 7.4 multicycle co-execution hazard); compiler can't see
  // inside the asm so we provide it ourselves.
  asm volatile("v_tanh_f32 %0, %1\n\tv_nop" : "=v"(r) : "v"(x));
  return r;
}
#endif
// sigmoid(x) = 0.5*tanh(0.5*x) + 0.5  (1 TRANS + 2 VALU)
__device__ __forceinline__ float sigm(float x) {
  return fmaf(0.5f, fast_tanh(0.5f * x), 0.5f);
}
// sigmoid(x + b) with pre-halved bias hb = 0.5*b
__device__ __forceinline__ float sigm_b(float x, float hb) {
  return fmaf(0.5f, fast_tanh(fmaf(0.5f, x, hb)), 0.5f);
}

__global__ __launch_bounds__(TPB, 1)
void gru_fused_kernel(const float* __restrict__ x,
                      const float* __restrict__ Wih0, const float* __restrict__ Whh0,
                      const float* __restrict__ bih0, const float* __restrict__ bhh0,
                      const float* __restrict__ Wih1, const float* __restrict__ Whh1,
                      const float* __restrict__ bih1, const float* __restrict__ bhh1,
                      const float* __restrict__ gamma, const float* __restrict__ beta,
                      const float* __restrict__ Wfc, const float* __restrict__ bfc,
                      float* __restrict__ out)
{
  __shared__ Lds L;
  const int tid  = threadIdx.x;
  const int wave = tid >> 5;
  const int lane = tid & 31;
  const int lh   = lane & 15;
  const int hi   = lane >> 4;

  // ---- pack B matrices (B[k,n] = W[n*HID+k]) into WMMA fragment order ----
  for (int idx = tid; idx < NT*KC*FRAG_H; idx += TPB) {
    const int jj = idx & 15;
    const int ln = (idx >> 4) & 31;
    const int kc = (idx >> 9) & (KC - 1);
    const int nt = idx >> 10;
    const int n  = nt*16 + (ln & 15);
    const int koff = ((ln >= 16) ? 8 : 0) + ((jj < 8) ? jj : jj + 8);
    const int k  = kc*32 + koff;
    L.Bhh0[idx] = (_Float16)Whh0[n*HID + k];
    L.Bih1[idx] = (_Float16)Wih1[n*HID + k];
    L.Bhh1[idx] = (_Float16)Whh1[n*HID + k];
  }
  // layer-0 input weights + "ones trick": k==2 row carries fused biases
  for (int idx = tid; idx < NT*FRAG_H; idx += TPB) {
    const int jj = idx & 15;
    const int ln = (idx >> 4) & 31;
    const int nt = idx >> 9;
    const int n  = nt*16 + (ln & 15);
    const int koff = ((ln >= 16) ? 8 : 0) + ((jj < 8) ? jj : jj + 8);
    float v = 0.0f;
    if (koff < CH)       v = Wih0[n*CH + koff];
    else if (koff == CH) v = (nt < 8) ? (bih0[n] + bhh0[n]) : bih0[n];
    L.Bih0[idx] = (_Float16)v;
  }
  for (int i = lane; i < 16*HID; i += 32) {
    L.h0t[wave][i] = (_Float16)0.0f;
    L.h1t[wave][i] = (_Float16)0.0f;
  }
  __syncthreads();

  const int  tile = blockIdx.x * WAVES + wave;     // 0..827
  const long row0 = (long)tile * 16;

  // per-lane constants (column index = a*16 + lh); r/z biases pre-halved
  float bhn0[4], hbr1[4], hbz1[4], bi1[4], bhn1[4], gam4[4], bet4[4], wf4[4];
#pragma unroll
  for (int a = 0; a < 4; ++a) {
    const int gr = a*16 + lh, gz = (a+4)*16 + lh, gn = (a+8)*16 + lh;
    bhn0[a] = bhh0[gn];
    hbr1[a] = 0.5f * (bih1[gr] + bhh1[gr]);
    hbz1[a] = 0.5f * (bih1[gz] + bhh1[gz]);
    bi1[a]  = bih1[gn];
    bhn1[a] = bhh1[gn];
    gam4[a] = gamma[a*16 + lh];
    bet4[a] = beta[a*16 + lh];
    wf4[a]  = Wfc[a*16 + lh];
  }
  const float bfcv = bfc[0];
  const v8f kZero = {};

  float h0c[4][8], h1c[4][8];
#pragma unroll
  for (int a = 0; a < 4; ++a)
#pragma unroll
    for (int j = 0; j < 8; ++j) { h0c[a][j] = 0.0f; h1c[a][j] = 0.0f; }

  _Float16* h0b = L.h0t[wave];
  _Float16* h1b = L.h1t[wave];
  const float* xrow = x + (row0 + lh) * (long)(SEQ*CH);

  // software-pipelined x load: xc holds x[t], xn prefetches x[t+1]
  float2 xc = *(const float2*)xrow;

  for (int t = 0; t < SEQ; ++t) {
    float2 xn = xc;
    if (t + 1 < SEQ) xn = *(const float2*)(xrow + (t+1)*2);

    // A-fragments of previous hidden states (DS ops are in-order per wave)
    v16h a00 = loadA(h0b, 0, lh, hi);
    v16h a01 = loadA(h0b, 1, lh, hi);
    v16h a10 = loadA(h1b, 0, lh, hi);
    v16h a11 = loadA(h1b, 1, lh, hi);
    asm volatile("" ::: "memory");

    // x fragment: lanes 0..15 carry K0=x0, K1=x1, K2=1 (bias row)
    v16h ax;
#pragma unroll
    for (int i = 0; i < 16; ++i) ax[i] = (_Float16)0.0f;
    ax[0] = (hi == 0) ? (_Float16)xc.x  : (_Float16)0.0f;
    ax[1] = (hi == 0) ? (_Float16)xc.y  : (_Float16)0.0f;
    ax[2] = (hi == 0) ? (_Float16)1.0f  : (_Float16)0.0f;

    // -------- layer 0 --------
#pragma unroll
    for (int ht = 0; ht < 4; ++ht) {
      const int gr = ht, gz = ht + 4, gn = ht + 8;
      v8f accr = wmma16(ax,  ldsB(&L.Bih0[(gr*32 + lane)*16]), kZero);
      accr = wmma16(a00, ldsB(&L.Bhh0[((gr*KC + 0)*32 + lane)*16]), accr);
      accr = wmma16(a01, ldsB(&L.Bhh0[((gr*KC + 1)*32 + lane)*16]), accr);
      v8f accz = wmma16(ax,  ldsB(&L.Bih0[(gz*32 + lane)*16]), kZero);
      accz = wmma16(a00, ldsB(&L.Bhh0[((gz*KC + 0)*32 + lane)*16]), accz);
      accz = wmma16(a01, ldsB(&L.Bhh0[((gz*KC + 1)*32 + lane)*16]), accz);
      v8f acci = wmma16(ax,  ldsB(&L.Bih0[(gn*32 + lane)*16]), kZero);
      v8f acch = wmma16(a00, ldsB(&L.Bhh0[((gn*KC + 0)*32 + lane)*16]), kZero);
      acch = wmma16(a01, ldsB(&L.Bhh0[((gn*KC + 1)*32 + lane)*16]), acch);
#pragma unroll
      for (int j = 0; j < 8; ++j) {
        const float r  = sigm(accr[j]);
        const float z  = sigm(accz[j]);
        const float nn = fast_tanh(fmaf(r, acch[j] + bhn0[ht], acci[j]));
        const float hv = fmaf(z, h0c[ht][j] - nn, nn);
        h0c[ht][j] = hv;
        h0b[(j + 8*hi)*HID + ht*16 + lh] = (_Float16)hv;
      }
    }
    asm volatile("" ::: "memory");

    // new layer-0 hidden -> A-fragments (input to layer 1 at this timestep)
    v16h b00 = loadA(h0b, 0, lh, hi);
    v16h b01 = loadA(h0b, 1, lh, hi);
    asm volatile("" ::: "memory");

    // -------- layer 1 --------
#pragma unroll
    for (int ht = 0; ht < 4; ++ht) {
      const int gr = ht, gz = ht + 4, gn = ht + 8;
      v8f accr = wmma16(b00, ldsB(&L.Bih1[((gr*KC + 0)*32 + lane)*16]), kZero);
      accr = wmma16(b01, ldsB(&L.Bih1[((gr*KC + 1)*32 + lane)*16]), accr);
      accr = wmma16(a10, ldsB(&L.Bhh1[((gr*KC + 0)*32 + lane)*16]), accr);
      accr = wmma16(a11, ldsB(&L.Bhh1[((gr*KC + 1)*32 + lane)*16]), accr);
      v8f accz = wmma16(b00, ldsB(&L.Bih1[((gz*KC + 0)*32 + lane)*16]), kZero);
      accz = wmma16(b01, ldsB(&L.Bih1[((gz*KC + 1)*32 + lane)*16]), accz);
      accz = wmma16(a10, ldsB(&L.Bhh1[((gz*KC + 0)*32 + lane)*16]), accz);
      accz = wmma16(a11, ldsB(&L.Bhh1[((gz*KC + 1)*32 + lane)*16]), accz);
      v8f acci = wmma16(b00, ldsB(&L.Bih1[((gn*KC + 0)*32 + lane)*16]), kZero);
      acci = wmma16(b01, ldsB(&L.Bih1[((gn*KC + 1)*32 + lane)*16]), acci);
      v8f acch = wmma16(a10, ldsB(&L.Bhh1[((gn*KC + 0)*32 + lane)*16]), kZero);
      acch = wmma16(a11, ldsB(&L.Bhh1[((gn*KC + 1)*32 + lane)*16]), acch);
#pragma unroll
      for (int j = 0; j < 8; ++j) {
        const float r  = sigm_b(accr[j], hbr1[ht]);
        const float z  = sigm_b(accz[j], hbz1[ht]);
        const float nn = fast_tanh(fmaf(r, acch[j] + bhn1[ht], acci[j] + bi1[ht]));
        const float hv = fmaf(z, h1c[ht][j] - nn, nn);
        h1c[ht][j] = hv;
        h1b[(j + 8*hi)*HID + ht*16 + lh] = (_Float16)hv;
      }
    }
    asm volatile("" ::: "memory");

    // -------- tail: LayerNorm over HID + FC (last PRE steps) --------
    if (t >= SEQ - PRE) {
      const int tp = t - (SEQ - PRE);
#pragma unroll
      for (int j = 0; j < 8; ++j) {
        float s = h1c[0][j] + h1c[1][j] + h1c[2][j] + h1c[3][j];
#pragma unroll
        for (int m = 1; m < 16; m <<= 1) s += __shfl_xor(s, m, 16);
        const float mu = s * (1.0f / HID);
        float v = 0.0f;
#pragma unroll
        for (int a = 0; a < 4; ++a) { const float d = h1c[a][j] - mu; v += d*d; }
#pragma unroll
        for (int m = 1; m < 16; m <<= 1) v += __shfl_xor(v, m, 16);
        const float rs = rsqrtf(v * (1.0f / HID) + 1e-5f);
        float acc = 0.0f;
#pragma unroll
        for (int a = 0; a < 4; ++a) {
          const float nv = (h1c[a][j] - mu) * rs * gam4[a] + bet4[a];
          acc += nv * wf4[a];
        }
#pragma unroll
        for (int m = 1; m < 16; m <<= 1) acc += __shfl_xor(acc, m, 16);
        if (lh == 0) {
          const long r  = row0 + j + 8*hi;
          const int  bb = (int)(r / NODES);
          const int  nn = (int)(r % NODES);
          out[((long)(bb*PRE + tp))*NODES + nn] = acc + bfcv;
        }
      }
    }
    xc = xn;
  }
}

extern "C" void kernel_launch(void* const* d_in, const int* in_sizes, int n_in,
                              void* d_out, int out_size, void* d_ws, size_t ws_size,
                              hipStream_t stream) {
  (void)in_sizes; (void)n_in; (void)out_size; (void)d_ws; (void)ws_size;
  const float* x     = (const float*)d_in[0];
  const float* Wih0  = (const float*)d_in[1];
  const float* Whh0  = (const float*)d_in[2];
  const float* bih0v = (const float*)d_in[3];
  const float* bhh0v = (const float*)d_in[4];
  const float* Wih1  = (const float*)d_in[5];
  const float* Whh1  = (const float*)d_in[6];
  const float* bih1v = (const float*)d_in[7];
  const float* bhh1v = (const float*)d_in[8];
  const float* gam   = (const float*)d_in[9];
  const float* bet   = (const float*)d_in[10];
  const float* Wfc   = (const float*)d_in[11];
  const float* bfcv  = (const float*)d_in[12];
  float* out = (float*)d_out;

  const int grid = ROWS / (WAVES * 16);   // 13248 / 64 = 207 blocks
  gru_fused_kernel<<<grid, TPB, 0, stream>>>(x, Wih0, Whh0, bih0v, bhh0v,
                                             Wih1, Whh1, bih1v, bhh1v,
                                             gam, bet, Wfc, bfcv, out);
}
